// Conv_ST_54254026883270
// MI455X (gfx1250) — matrix operand order repaired and checked by reading, not compiled, not run on hardware
//
#include <hip/hip_runtime.h>
#include <hip/hip_bf16.h>

// ---------------- types ----------------
typedef __bf16 v16bf __attribute__((ext_vector_type(16)));
typedef float  v8f   __attribute__((ext_vector_type(8)));
typedef unsigned short u16;
typedef int v4i_vs __attribute__((__vector_size__(16)));   // matches builtin proto

union FragV { uint4 q[2]; v16bf v; };

// problem constants
#define BB 8
#define TT 12
#define NN 2048
#define CC 64
#define DD 192          // KT*CC
#define NCOL 256        // stage-1 GEMM columns (o,j)
#define MTOT 96         // B*T
#define ZCOLS 12288     // MTOT*128
#define ASTRIDE 40      // 32 + 8 pad (halves) -> bank-conflict-free

// ---------------- CDNA5 async global->LDS staging (guarded) ----------------
#if defined(__HIP_DEVICE_COMPILE__) && defined(__gfx1250__) && \
    __has_builtin(__builtin_amdgcn_global_load_async_to_lds_b128)
#define USE_ASYNC 1
#else
#define USE_ASYNC 0
#endif

#if USE_ASYNC
#define ASYNC_CP16(gsrc, ldst)                                                  \
  __builtin_amdgcn_global_load_async_to_lds_b128(                               \
      (__attribute__((address_space(1))) v4i_vs*)(gsrc),                        \
      (__attribute__((address_space(3))) v4i_vs*)(ldst), 0, 0)
#if __has_builtin(__builtin_amdgcn_s_wait_asynccnt)
#define WAIT_ASYNC(n) __builtin_amdgcn_s_wait_asynccnt(n)
#else
#define WAIT_ASYNC(n) asm volatile("s_wait_asynccnt %0" ::"i"(n) : "memory")
#endif
#else
#define ASYNC_CP16(gsrc, ldst) (*(uint4*)(ldst) = *(const uint4*)(gsrc))
#define WAIT_ASYNC(n)
#endif

// ---------------- helpers ----------------
__device__ __forceinline__ u16 f2bf(float x) {
  unsigned int u = __float_as_uint(x);
  unsigned int r = u + 0x7FFFu + ((u >> 16) & 1u);
  return (u16)(r >> 16);
}
__device__ __forceinline__ unsigned int pk2(float a, float b) {
  return (unsigned int)f2bf(a) | ((unsigned int)f2bf(b) << 16);
}
__device__ __forceinline__ uint4 pack8(float4 a, float4 b) {
  uint4 r; r.x = pk2(a.x, a.y); r.y = pk2(a.z, a.w);
  r.z = pk2(b.x, b.y); r.w = pk2(b.z, b.w); return r;
}
__device__ __forceinline__ uint4 pack8v(v8f v) {
  uint4 r; r.x = pk2(v[0], v[1]); r.y = pk2(v[2], v[3]);
  r.z = pk2(v[4], v[5]); r.w = pk2(v[6], v[7]); return r;
}

// ---------------- K0: supports -> St[o][k][i] (bf16, transposed) ----------------
__global__ __launch_bounds__(256) void k_prep_supports(const float* __restrict__ sup,
                                                       u16* __restrict__ St) {
  __shared__ float tile[32][33];
  const int o = blockIdx.z;
  const int i0 = blockIdx.x * 32;
  const int k0 = blockIdx.y * 32;
  const int tx = threadIdx.x & 31;
  const int ty = threadIdx.x >> 5;
#pragma unroll
  for (int q = 0; q < 32; q += 8)
    tile[ty + q][tx] = sup[(size_t)(i0 + ty + q) * (2 * NN) + (size_t)o * NN + k0 + tx];
  __syncthreads();
#pragma unroll
  for (int q = 0; q < 32; q += 8)
    St[(size_t)o * NN * NN + (size_t)(k0 + ty + q) * NN + i0 + tx] = f2bf(tile[tx][ty + q]);
}

// ---------------- K1: weights -> W2cm[col][d] bf16, col = o*128 + j ----------------
__global__ __launch_bounds__(256) void k_prep_weights(const float* __restrict__ w,
                                                      u16* __restrict__ W2) {
  int e = blockIdx.x * 256 + threadIdx.x;
  if (e < DD * NCOL) {
    int col = e / DD, d = e % DD;
    int o = col >> 7, j = col & 127;
    W2[(size_t)col * DD + d] = f2bf(w[(size_t)(d * 2 + o) * 128 + j]);
  }
}

// ---------------- K2: stage-1 GEMM  z = x @ W  (WMMA bf16) ----------------
// rows R = m*2048 + i  (64 per block), cols = 256 (o,j); K = 192
// output Zcm[o][m*128+j][i] (column-major over i), bf16
__global__ __launch_bounds__(256) void k_stage1(const float* __restrict__ inputs,
                                                const u16* __restrict__ W2,
                                                u16* __restrict__ Zcm) {
  __shared__ u16 A1[64 * ASTRIDE];
  __shared__ u16 B1[NCOL * ASTRIDE];

  const int tid  = threadIdx.x;
  const int lane = tid & 31;
  const int w    = tid >> 5;
  const int wr0  = (w & 1) * 32;
  const int wc0  = (w >> 1) * 64;
  const int rA   = lane & 15;
  const int hiA  = (lane >> 4);

  const int R0    = blockIdx.x * 64;
  const int m_    = R0 >> 11;
  const int iBase = R0 & (NN - 1);
  const int tl    = m_ % TT;

  v8f acc[2][4];
#pragma unroll
  for (int rt = 0; rt < 2; ++rt)
#pragma unroll
    for (int ct = 0; ct < 4; ++ct) acc[rt][ct] = (v8f){0.f,0.f,0.f,0.f,0.f,0.f,0.f,0.f};

  for (int ks = 0; ks < 6; ++ks) {
    const int ktap = ks >> 1;
    const int c0   = (ks & 1) * 32;
    __syncthreads();
    {   // A tile: causal shift + f32->bf16
      const int r = tid >> 2, p = tid & 3;
      const int ts = tl + ktap - 2;
      uint4 val = {0u, 0u, 0u, 0u};
      if (ts >= 0) {
        const float* src = inputs +
            (((size_t)(m_ + ktap - 2) * NN + (iBase + r)) * CC + c0 + p * 8);
        float4 f0 = *(const float4*)src;
        float4 f1 = *(const float4*)(src + 4);
        val = pack8(f0, f1);
      }
      *(uint4*)&A1[r * ASTRIDE + p * 8] = val;
    }
#pragma unroll
    for (int q = 0; q < 4; ++q) {   // B tile from W2cm
      int c = tid + q * 256;
      int col = c >> 2, p = c & 3;
      *(uint4*)&B1[col * ASTRIDE + p * 8] =
          *(const uint4*)(W2 + (size_t)col * DD + ks * 32 + p * 8);
    }
    __syncthreads();

    FragV a[2], b[4];
#pragma unroll
    for (int rt = 0; rt < 2; ++rt) {
      int row = wr0 + rt * 16 + rA;
      int base = hiA * 16;
      a[rt].q[0] = *(const uint4*)&A1[row * ASTRIDE + base];
      a[rt].q[1] = *(const uint4*)&A1[row * ASTRIDE + base + 8];
    }
#pragma unroll
    for (int ct = 0; ct < 4; ++ct) {
      int col = wc0 + ct * 16 + rA;
      int kb  = hiA * 16;
      b[ct].q[0] = *(const uint4*)&B1[col * ASTRIDE + kb];
      b[ct].q[1] = *(const uint4*)&B1[col * ASTRIDE + kb + 8];
    }
#pragma unroll
    for (int rt = 0; rt < 2; ++rt)
#pragma unroll
      for (int ct = 0; ct < 4; ++ct)
        acc[rt][ct] = __builtin_amdgcn_wmma_f32_16x16x32_bf16(
            false, a[rt].v, false, b[ct].v, (short)0, acc[rt][ct], false, false);
  }

#pragma unroll
  for (int rt = 0; rt < 2; ++rt) {
#pragma unroll
    for (int ct = 0; ct < 4; ++ct) {
      int col256 = wc0 + ct * 16 + rA;
      int o = col256 >> 7, j = col256 & 127;
      int rloc = wr0 + rt * 16 + ((lane & 16) ? 8 : 0);
      u16* dst = Zcm + ((size_t)o * ZCOLS + (size_t)m_ * 128 + j) * NN + iBase + rloc;
      *(uint4*)dst = pack8v(acc[rt][ct]);
    }
  }
}

// ---------------- K3: stage-2 GEMM + GLU (WMMA bf16, async double-buffered) ----
// Conv[k, m*128+j] = sum_o St_o[k,:] . Zcm_o[m*128+j,:]   (K = 2048 per o)
__global__ __launch_bounds__(256) void k_stage2(const u16* __restrict__ St,
                                                const u16* __restrict__ Zcm,
                                                const float* __restrict__ inputs,
                                                const float* __restrict__ bias,
                                                float* __restrict__ out) {
  // staging buffers (double-buffered) overlaid with GLU gate-exchange buffer
  __shared__ __align__(16) union S2 {
    struct { u16 A[2][128 * ASTRIDE]; u16 B[2][128 * ASTRIDE]; } st;  // 40960 B
    float gate[128 * 68];                                             // 34816 B
  } sm;

  const int tid  = threadIdx.x;
  const int lane = tid & 31;
  const int w    = tid >> 5;
  const int wr0  = (w & 3) * 32;
  const int wc0  = (w >> 2) * 64;
  const int rA   = lane & 15;
  const int hiA  = (lane >> 4);

  const int m  = blockIdx.x;
  const int k0 = blockIdx.y * 128;
  const int n0 = m * 128;

  // per-thread staging coordinates: 2 chunks of A + 2 of B per K-step
  const int r0c = tid >> 2,          p0c = tid & 3;
  const int r1c = (tid + 256) >> 2,  p1c = tid & 3;

  const int NIT = 128;   // 2 (order) x 64 K-steps of 32

  auto issue = [&](int it, int buf) {
    const int o  = it >> 6;
    const int kk = (it & 63) * 32;
    const u16* Sg = St  + (size_t)o * NN * NN    + (size_t)k0 * NN + kk;
    const u16* Zg = Zcm + (size_t)o * ZCOLS * NN + (size_t)n0 * NN + kk;
    ASYNC_CP16(Sg + (size_t)r0c * NN + p0c * 8, &sm.st.A[buf][r0c * ASTRIDE + p0c * 8]);
    ASYNC_CP16(Zg + (size_t)r0c * NN + p0c * 8, &sm.st.B[buf][r0c * ASTRIDE + p0c * 8]);
    ASYNC_CP16(Sg + (size_t)r1c * NN + p1c * 8, &sm.st.A[buf][r1c * ASTRIDE + p1c * 8]);
    ASYNC_CP16(Zg + (size_t)r1c * NN + p1c * 8, &sm.st.B[buf][r1c * ASTRIDE + p1c * 8]);
  };

  v8f acc[2][4];
#pragma unroll
  for (int rt = 0; rt < 2; ++rt)
#pragma unroll
    for (int ct = 0; ct < 4; ++ct) acc[rt][ct] = (v8f){0.f,0.f,0.f,0.f,0.f,0.f,0.f,0.f};

  issue(0, 0);
  for (int it = 0; it < NIT; ++it) {
    const int cur = it & 1;
    if (it + 1 < NIT) {               // prefetch next K-slice into other buffer
      issue(it + 1, cur ^ 1);
      WAIT_ASYNC(4);                  // oldest batch (this slice) resident
    } else {
      WAIT_ASYNC(0);                  // drain
    }
    __syncthreads();

    const u16* Ab = sm.st.A[cur];
    const u16* Bb = sm.st.B[cur];
    FragV a[2], b[4];
#pragma unroll
    for (int rt = 0; rt < 2; ++rt) {
      int row = wr0 + rt * 16 + rA;
      int base = hiA * 16;
      a[rt].q[0] = *(const uint4*)&Ab[row * ASTRIDE + base];
      a[rt].q[1] = *(const uint4*)&Ab[row * ASTRIDE + base + 8];
    }
#pragma unroll
    for (int ct = 0; ct < 4; ++ct) {
      int col = wc0 + ct * 16 + rA;
      int kb  = hiA * 16;
      b[ct].q[0] = *(const uint4*)&Bb[col * ASTRIDE + kb];
      b[ct].q[1] = *(const uint4*)&Bb[col * ASTRIDE + kb + 8];
    }
#pragma unroll
    for (int rt = 0; rt < 2; ++rt)
#pragma unroll
      for (int ct = 0; ct < 4; ++ct)
        acc[rt][ct] = __builtin_amdgcn_wmma_f32_16x16x32_bf16(
            false, a[rt].v, false, b[ct].v, (short)0, acc[rt][ct], false, false);
    __syncthreads();                  // all reads done before buffer reuse
  }

  // ---- GLU epilogue: gate half (cols 64..127) -> LDS, lhs half writes out ----
  if (wc0 == 64) {
#pragma unroll
    for (int rt = 0; rt < 2; ++rt)
#pragma unroll
      for (int ct = 0; ct < 4; ++ct) {
        int j2 = ct * 16 + rA;
#pragma unroll
        for (int g = 0; g < 8; ++g) {
          int row = wr0 + rt * 16 + g + ((lane & 16) ? 8 : 0);
          sm.gate[row * 68 + j2] = acc[rt][ct][g] + bias[64 + j2];
        }
      }
  }
  __syncthreads();
  if (wc0 == 0) {
#pragma unroll
    for (int rt = 0; rt < 2; ++rt)
#pragma unroll
      for (int ct = 0; ct < 4; ++ct) {
        int j = ct * 16 + rA;
#pragma unroll
        for (int g = 0; g < 8; ++g) {
          int row = wr0 + rt * 16 + g + ((lane & 16) ? 8 : 0);
          size_t idx = ((size_t)m * NN + (k0 + row)) * CC + j;
          float lhs = acc[rt][ct][g] + bias[j] + inputs[idx];
          float ga  = sm.gate[row * 68 + j];
          out[idx] = lhs * (1.0f / (1.0f + __expf(-ga)));
        }
      }
  }
}

// ---------------- launch ----------------
extern "C" void kernel_launch(void* const* d_in, const int* in_sizes, int n_in,
                              void* d_out, int out_size, void* d_ws, size_t ws_size,
                              hipStream_t stream) {
  const float* inputs   = (const float*)d_in[0];   // [8,12,2048,64]
  const float* supports = (const float*)d_in[1];   // [2048,4096]
  const float* weights  = (const float*)d_in[2];   // [384,128]
  const float* biases   = (const float*)d_in[3];   // [128]
  float* out = (float*)d_out;

  const size_t stBytes = (size_t)2 * NN * NN * 2;            // 16.78 MB
  const size_t w2Bytes = (size_t)DD * NCOL * 2;              // 96 KB
  u16* St  = (u16*)d_ws;
  u16* W2  = (u16*)((char*)d_ws + stBytes);
  u16* Zcm = (u16*)((char*)d_ws + stBytes + ((w2Bytes + 255) & ~(size_t)255));

  k_prep_supports<<<dim3(NN / 32, NN / 32, 2), 256, 0, stream>>>(supports, St);
  k_prep_weights<<<(DD * NCOL + 255) / 256, 256, 0, stream>>>(weights, W2);
  k_stage1<<<(MTOT * NN) / 64, 256, 0, stream>>>(inputs, W2, Zcm);
  k_stage2<<<dim3(MTOT, NN / 128), 256, 0, stream>>>(St, Zcm, inputs, biases, out);
}
